// Stack_83253646065813
// MI455X (gfx1250) — matrix operand order repaired and checked by reading, not compiled, not run on hardware
//
#include <hip/hip_runtime.h>
#include <hip/hip_bf16.h>
#include <hip/hip_fp16.h>

// CDNA5 / gfx1250, wave32.
typedef __attribute__((ext_vector_type(16))) _Float16 v16h;
typedef __attribute__((ext_vector_type(8)))  _Float16 v8h;
typedef __attribute__((ext_vector_type(8)))  float    v8f;

#define T_STEPS 128
#define DIN     768
#define HID     768
#define NOUT    768
#define NLAYERS 3
#define NROWS   (NLAYERS * HID)   // 2304 independent row-scans
#define CHUNK   (DIN / 32)        // 24 elements per lane (wave32)

// ---------------------------------------------------------------------------
// Phase 1: per-row sequential scan over T. One wave32 per row; W/P/eta rows
// live entirely in VGPRs (24 each per lane). x_t is staged into LDS once per
// block (double-buffered, one s_barrier per step). Emits h_t as f16 into H.
// ---------------------------------------------------------------------------
__global__ __launch_bounds__(256) void plastic_scan_kernel(
    const float* __restrict__ x_seq, const float* __restrict__ W,
    const float* __restrict__ b,     const float* __restrict__ lam,
    const float* __restrict__ eta0,  const float* __restrict__ eta1,
    const float* __restrict__ eta2,  const float* __restrict__ P_init,
    _Float16* __restrict__ H)
{
    __shared__ float xs[2][DIN];

    const int lane  = threadIdx.x & 31;
    const int wave  = threadIdx.x >> 5;
    const int r     = blockIdx.x * 8 + wave;       // global row 0..2303
    const int layer = r / HID;                     // 0,1,2
    const int rl    = r - layer * HID;             // row within layer
    const bool isM  = (layer == 1);                // TYPES = (A, M, A)

    // Per-layer constants
    const float lam_i  = lam[layer];
    const float decay  = 1.0f / (1.0f + __expf(-lam_i));   // sigmoid(lambda)
    const float b_r    = b[r];
    const float eta_sc = eta1[0];                          // scalar eta (layer 1)

    // Row-resident state in registers (wave32: j = lane + 32*q)
    float Wreg[CHUNK], Preg[CHUNK], Ereg[CHUNK];
    {
        const float* wrow = W      + (size_t)r  * DIN;
        const float* prow = P_init + (size_t)r  * DIN;
        const float* erow = (layer == 2) ? (eta2 + (size_t)rl * DIN)
                                         : (eta0 + (size_t)rl * DIN); // valid addr even for layer 1
        #pragma unroll
        for (int q = 0; q < CHUNK; ++q) {
            const int j = lane + 32 * q;
            Wreg[q] = wrow[j];
            Preg[q] = prow[j];
            Ereg[q] = isM ? eta_sc : erow[j];
        }
    }

    // Preload x_0 into LDS buffer 0 (cooperative, coalesced)
    for (int k = threadIdx.x; k < DIN; k += 256) xs[0][k] = x_seq[k];
    __syncthreads();

    for (int t = 0; t < T_STEPS; ++t) {
        // Snapshot x_t from LDS into registers (conflict-free: stride-32 per lane)
        const float* xv = xs[t & 1];
        float xr[CHUNK];
        #pragma unroll
        for (int q = 0; q < CHUNK; ++q) xr[q] = xv[lane + 32 * q];

        // Stage x_{t+1} into the other LDS buffer; prefetch x_{t+2} into GL2
        if (t + 1 < T_STEPS) {
            const float* xn = x_seq + (size_t)(t + 1) * DIN;
            float* dst = xs[(t + 1) & 1];
            for (int k = threadIdx.x; k < DIN; k += 256) dst[k] = xn[k];
        }
        if (t + 2 < T_STEPS)
            __builtin_prefetch(x_seq + (size_t)(t + 2) * DIN, 0, 3);
        __syncthreads();   // single barrier/step covers both buffers

        // pre = effective_row . x  (+ b)
        float acc = 0.0f;
        #pragma unroll
        for (int q = 0; q < CHUNK; ++q) {
            const float w = Wreg[q], p = Preg[q];
            const float eff = isM ? (w * (p + 1.0f))   // (W*P)@x + W@x
                                  : (w + p);           // (W+P)@x
            acc = fmaf(eff, xr[q], acc);
        }
        // wave32 butterfly reduction — all lanes get the full dot product
        #pragma unroll
        for (int off = 16; off >= 1; off >>= 1)
            acc += __shfl_xor(acc, off, 32);

        const float h = 1.0f / (1.0f + __expf(-(acc + b_r)));
        if (lane == 0)
            H[(size_t)t * NROWS + r] = (_Float16)h;

        // P <- decay*P + eta ∘ (h * x)   (row-local, stays in VGPRs)
        #pragma unroll
        for (int q = 0; q < CHUNK; ++q)
            Preg[q] = fmaf(Ereg[q] * h, xr[q], decay * Preg[q]);
    }
}

// ---------------------------------------------------------------------------
// Phase 2: Y = sigmoid(H(128x2304) * w_final^T(2304x768) + b_final) via
// v_wmma_f32_16x16x32_f16. One wave per 16x16 output tile; K-loop of 72.
// Fragment layouts used here (contiguous k order, same for A and B):
//   A(16x32 f16): lane l holds M=l&15; half e -> K=base16+e, base16 = 16*(l>=16).
//   B(32x16 f16): lane l holds N=l&15; half e -> K=base16+e, base16 = 16*(l>=16).
//   C/D (v8f):    vgpr j -> row M = j + 8*(l>=16), col N = l&15.
// ---------------------------------------------------------------------------
__global__ __launch_bounds__(256) void head_gemm_kernel(
    const _Float16* __restrict__ H, const float* __restrict__ w_final,
    const float* __restrict__ b_final, float* __restrict__ out)
{
    const int lane   = threadIdx.x & 31;
    const int mtile  = threadIdx.x >> 5;   // 8 waves cover all 8 M-tiles (T=128)
    const int ntile  = blockIdx.x;         // 48 blocks cover N=768
    const int hi     = lane >> 4;          // 0 | 1
    const int l15    = lane & 15;
    const int base8  = hi * 8;
    const int base16 = hi * 16;

    const int m = mtile * 16 + l15;        // A row (time index) this lane loads
    const int n = ntile * 16 + l15;        // B col (output unit) this lane loads

    const _Float16* __restrict__ arow = H       + (size_t)m * NROWS;
    const float*    __restrict__ brow = w_final + (size_t)n * NROWS; // w_final[n][k], k fast

    v8f acc = {};
    for (int kb = 0; kb < NROWS; kb += 32) {
        // A fragment: two contiguous 16B runs of H (f16, row-major over k)
        const v8h alo = *(const v8h*)(arow + kb + base16);
        const v8h ahi = *(const v8h*)(arow + kb + base16 + 8);
        v16h a;
        #pragma unroll
        for (int e = 0; e < 8; ++e) { a[e] = alo[e]; a[e + 8] = ahi[e]; }

        // B fragment: 16 contiguous f32 of w_final row n, converted to f16
        const float4* bp = (const float4*)(brow + kb + base16);
        v16h bf;
        #pragma unroll
        for (int v = 0; v < 4; ++v) {
            const float4 f = bp[v];
            bf[4 * v + 0] = (_Float16)f.x;  bf[4 * v + 1] = (_Float16)f.y;
            bf[4 * v + 2] = (_Float16)f.z;  bf[4 * v + 3] = (_Float16)f.w;
        }

        // D = A x B + C, f32 accumulate
        acc = __builtin_amdgcn_wmma_f32_16x16x32_f16(
                  /*neg_a=*/false, a, /*neg_b=*/false, bf,
                  /*c_mod=*/(short)0, acc, /*reuse_a=*/false, /*reuse_b=*/false);
    }

    // Epilogue: bias + sigmoid, scatter per C/D layout
    const int col = ntile * 16 + l15;
    const float bfin = b_final[col];
    #pragma unroll
    for (int j = 0; j < 8; ++j) {
        const int row = mtile * 16 + j + hi * 8;
        const float v = acc[j] + bfin;
        out[(size_t)row * NOUT + col] = 1.0f / (1.0f + __expf(-v));
    }
}

// ---------------------------------------------------------------------------
// Inputs (setup_inputs order): x_seq, W, b, lam, eta0, eta1(scalar), eta2,
// w_final, b_final, P_init. Output: y_seq (128x768 f32).
// Workspace: H = 128 x 2304 f16 (589,824 bytes) at d_ws.
// ---------------------------------------------------------------------------
extern "C" void kernel_launch(void* const* d_in, const int* in_sizes, int n_in,
                              void* d_out, int out_size, void* d_ws, size_t ws_size,
                              hipStream_t stream)
{
    const float* x_seq   = (const float*)d_in[0];
    const float* W       = (const float*)d_in[1];
    const float* b       = (const float*)d_in[2];
    const float* lam     = (const float*)d_in[3];
    const float* eta0    = (const float*)d_in[4];
    const float* eta1    = (const float*)d_in[5];
    const float* eta2    = (const float*)d_in[6];
    const float* w_final = (const float*)d_in[7];
    const float* b_final = (const float*)d_in[8];
    const float* P_init  = (const float*)d_in[9];

    float*    out = (float*)d_out;
    _Float16* H   = (_Float16*)d_ws;

    // 2304 rows, 8 wave32 rows per 256-thread block -> 288 blocks
    plastic_scan_kernel<<<NROWS / 8, 256, 0, stream>>>(
        x_seq, W, b, lam, eta0, eta1, eta2, P_init, H);

    // 48 blocks x 8 waves = 384 tiles = (128/16) x (768/16)
    head_gemm_kernel<<<NOUT / 16, 256, 0, stream>>>(H, w_final, b_final, out);
}
